// GraphDotProductLayer_28724741276293
// MI455X (gfx1250) — compile-verified, matrix operand order checked by reading, not run on hardware
//
#include <hip/hip_runtime.h>
#include <hip/hip_bf16.h>

typedef _Float16 half_t;
typedef __attribute__((ext_vector_type(16))) _Float16 v16h;
typedef __attribute__((ext_vector_type(8)))  _Float16 v8h;
typedef __attribute__((ext_vector_type(8)))  float    v8f;
typedef __attribute__((ext_vector_type(2)))  float    v2f;
typedef __attribute__((ext_vector_type(4)))  unsigned int u32x4;
typedef __attribute__((ext_vector_type(8)))  unsigned int u32x8;

#define B_   64
#define T_   512
#define D_   300
#define DM_  600
#define H_   10
#define KP_  608   // padded DM (mult of 32)
#define DHP_ 64    // padded head dim
#define CT_  640   // padded ctx cols (H_*DHP_)
#define NP_  320   // padded out cols

// ---- WMMA helpers (CDNA5 wave32 layouts) ------------------------------------
// A-operand f16 16x32: lane l: m=l&15, half=l>>4; elems 0..7 -> k=half*8+e,
// elems 8..15 -> k=16+half*8+(e-8). B-operand mirrors A with n=l&15 over rows
// of the (pre)transposed matrix, so both load two contiguous 16B chunks.
__device__ __forceinline__ v16h load_tile16(const half_t* base, int ld, int k0) {
  int lane = threadIdx.x & 31;
  int m = lane & 15, hf = lane >> 4;
  const half_t* p = base + (size_t)m * ld + k0 + hf * 8;
  v8h a = *(const v8h*)p;
  v8h b = *(const v8h*)(p + 16);
  v16h r;
#pragma unroll
  for (int i = 0; i < 8; ++i) { r[i] = a[i]; r[i + 8] = b[i]; }
  return r;
}

__device__ __forceinline__ v8f wmma16(v16h a, v16h b, v8f c) {
  return __builtin_amdgcn_wmma_f32_16x16x32_f16(false, a, false, b, (short)0, c, false, false);
}

// f32 A-operand 16x4: lane l: m=l&15, half=l>>4 -> {A[m][k0+2h], A[m][k0+2h+1]}
__device__ __forceinline__ v2f load_tile4(const float* base, int ld, int k0) {
  int lane = threadIdx.x & 31;
  int m = lane & 15, hf = lane >> 4;
  const float* p = base + (size_t)m * ld + k0 + hf * 2;
  return *(const v2f*)p;
}

// One ASYNCcnt-tracked 16B copy global->LDS (per-lane addresses).
__device__ __forceinline__ void async_copy_b128(half_t* lds, const half_t* gptr) {
  unsigned dst = (unsigned)(size_t)lds;                     // low 32b = LDS offset
  unsigned long long src = (unsigned long long)(size_t)gptr;
  asm volatile("global_load_async_to_lds_b128 %0, %1, off" :: "v"(dst), "v"(src) : "memory");
}

__device__ __forceinline__ void wait_asynccnt0() {
  asm volatile("s_wait_asynccnt 0" ::: "memory");
}

// TDM: one 2-D tensor_load_to_lds of `rows` x `cols` f16 (row stride = cols,
// i.e. contiguous tile) from `gptr` into LDS at `lds`. Descriptor per ISA §8.
__device__ __forceinline__ void tdm_load_2d_f16(half_t* lds, const half_t* gptr,
                                                unsigned cols, unsigned rows) {
  unsigned long long ga = (unsigned long long)(size_t)gptr;
  u32x4 g0 = { 1u,                                   // count=1, user descriptor
               (unsigned)(size_t)lds,                // lds_addr
               (unsigned)ga,                         // global_addr[31:0]
               (unsigned)((ga >> 32) & 0x01FFFFFFull) | 0x80000000u }; // addr[56:32]|type=2
  u32x8 g1 = { 1u << 16,                             // data_size=1 (2 bytes)
               (cols & 0xFFFFu) << 16,               // tensor_dim0[15:0]
               (rows & 0xFFFFu) << 16,               // dim0 hi=0 | tensor_dim1[15:0]
               (cols & 0xFFFFu) << 16,               // dim1 hi=0 | tile_dim0
               rows & 0xFFFFu,                       // tile_dim1 | tile_dim2=0
               cols,                                 // tensor_dim0_stride[31:0]
               0u, 0u };                             // stride hi, dim1_stride
  asm volatile("tensor_load_to_lds %0, %1" :: "s"(g0), "s"(g1) : "memory");
}

// ---- K0: convert + transpose + pad weights to f16 ---------------------------
__global__ void prep_weights(const float* __restrict__ Wq, const float* __restrict__ Wk,
                             const float* __restrict__ Wv, const float* __restrict__ Wo,
                             half_t* __restrict__ WTs, half_t* __restrict__ WopT) {
  const int NW = H_ * DHP_ * KP_;  // 389120
  int idx = blockIdx.x * blockDim.x + threadIdx.x;
  if (idx < 3 * NW) {
    int which = idx / NW, r = idx % NW;
    int row = r / KP_, k = r % KP_;
    int h = row >> 6, dp = row & 63;
    const float* W = (which == 0) ? Wq : (which == 1) ? Wk : Wv;
    float v = (dp < 60 && k < DM_) ? W[(size_t)k * DM_ + h * 60 + dp] : 0.0f;
    WTs[(size_t)which * NW + r] = (half_t)v;
  } else {
    int r = idx - 3 * NW;
    if (r < NP_ * CT_) {
      int n = r / CT_, c = r % CT_;
      int h = c >> 6, d = c & 63;
      float v = (n < D_ && d < 60) ? Wo[(size_t)(h * 60 + d) * D_ + n] : 0.0f;
      WopT[r] = (half_t)v;
    }
  }
}

// ---- K1: build ti = [x, asp] in f16, K-padded -------------------------------
__global__ void build_ti(const float* __restrict__ x, const int* __restrict__ aidx,
                         half_t* __restrict__ tiH) {
  int t = blockIdx.x, b = blockIdx.y;
  int i0 = aidx[b * 2], i1 = aidx[b * 2 + 1];
  const float* xb = x + (size_t)b * T_ * D_;
  half_t* dst = tiH + ((size_t)b * T_ + t) * KP_;
  for (int c = threadIdx.x; c < KP_; c += blockDim.x) {
    float v;
    if (c < D_) v = xb[(size_t)t * D_ + c];
    else if (c < DM_) { int d = c - D_; v = 0.5f * (xb[(size_t)i0 * D_ + d] + xb[(size_t)i1 * D_ + d]); }
    else v = 0.0f;
    dst[c] = (half_t)v;
  }
}

// ---- K2: QKV GEMM with LDS staging (TDM for weights, async for activations) -
// ti[64,608] x W^T[64,608] -> Qp/Kp [B,H,T,64], VT [B,H,64,T]
__global__ void qkv_gemm(const half_t* __restrict__ tiH, const half_t* __restrict__ WTs,
                         const float* __restrict__ bq, const float* __restrict__ bk,
                         const float* __restrict__ bv,
                         half_t* __restrict__ Qp, half_t* __restrict__ Kp,
                         half_t* __restrict__ VTl) {
  __shared__ __align__(16) half_t sA[64 * KP_];  // 76 KB activations
  __shared__ __align__(16) half_t sW[64 * KP_];  // 76 KB weight slice
  int b = blockIdx.z;
  int h = blockIdx.y % H_;
  int which = blockIdx.y / H_;
  int tid = threadIdx.x, w = tid >> 5;
  int t0b = blockIdx.x * 64;
  const half_t* Ag = tiH + ((size_t)b * T_ + t0b) * KP_;
  const half_t* Wg = WTs + (size_t)which * (H_ * DHP_ * KP_) + (size_t)h * DHP_ * KP_;

  if (w == 0)  // one TDM descriptor stages the whole 64x608 weight slice
    tdm_load_2d_f16(sW, Wg, KP_, 64);
  for (int i = tid * 8; i < 64 * KP_; i += 128 * 8)  // 38 x b128 per thread
    async_copy_b128(sA + i, Ag + i);
  wait_asynccnt0();
  if (w == 0) __builtin_amdgcn_s_wait_tensorcnt(0);
  __syncthreads();

  const half_t* A = sA + (size_t)(w * 16) * KP_;
  v8f acc[4] = {};
  for (int kc = 0; kc < KP_ / 32; ++kc) {
    int k0 = kc * 32;
    v16h a = load_tile16(A, KP_, k0);
#pragma unroll
    for (int j = 0; j < 4; ++j) {
      v16h bm = load_tile16(sW + (size_t)(j * 16) * KP_, KP_, k0);
      acc[j] = wmma16(a, bm, acc[j]);
    }
  }
  int lane = tid & 31, hf = lane >> 4, n = lane & 15;
  int t0 = t0b + w * 16;
  const float* bias = (which == 0) ? bq : (which == 1) ? bk : bv;
  const float invs = 0.12909944487358056f;  // 1/sqrt(60), folded into Q
#pragma unroll
  for (int j = 0; j < 4; ++j) {
    int dp = j * 16 + n;
    float bs = (dp < 60) ? bias[h * 60 + dp] : 0.0f;
#pragma unroll
    for (int r = 0; r < 8; ++r) {
      int m = r + 8 * hf;
      float v = acc[j][r] + bs;
      if (which == 0)
        Qp[(((size_t)(b * H_ + h)) * T_ + (t0 + m)) * DHP_ + dp] = (half_t)(v * invs);
      else if (which == 1)
        Kp[(((size_t)(b * H_ + h)) * T_ + (t0 + m)) * DHP_ + dp] = (half_t)v;
      else
        VTl[(((size_t)(b * H_ + h)) * DHP_ + dp) * T_ + (t0 + m)] = (half_t)v;
    }
  }
}

// ---- K3: fused attention: scores -> softmax -> ctx + head-mean -> mask ------
__global__ void attention(const half_t* __restrict__ Qp, const half_t* __restrict__ Kp,
                          const half_t* __restrict__ VTl, const int* __restrict__ adj,
                          half_t* __restrict__ ctxp, float* __restrict__ masked) {
  __shared__ __align__(16) float    sS[16 * T_];     // 32 KB scores/probs
  __shared__ __align__(16) _Float16 sP[16 * T_];     // 16 KB f16 probs
  __shared__ __align__(16) float    sMean[16 * T_];  // 32 KB head-mean acc
  __shared__ float sRed[128];
  int b = blockIdx.y;
  int q0 = blockIdx.x * 16;
  int tid = threadIdx.x;
  int w = tid >> 5, lane = tid & 31;
  int hf = lane >> 4, nn = lane & 15;
  int row = tid >> 3, seg = tid & 7;  // softmax: 8 threads per row

  for (int i = tid; i < 16 * T_; i += 128) sMean[i] = 0.0f;
  __syncthreads();

  for (int h = 0; h < H_; ++h) {
    // scores: S[16,512] = Q(16x64) . K^T, wave w handles k-tiles w,w+4,...
    const half_t* Qb = Qp + (((size_t)(b * H_ + h)) * T_ + q0) * DHP_;
    v16h qa0 = load_tile16(Qb, DHP_, 0);
    v16h qa1 = load_tile16(Qb, DHP_, 32);
    for (int nt = w; nt < T_ / 16; nt += 4) {
      const half_t* Kb = Kp + (((size_t)(b * H_ + h)) * T_ + nt * 16) * DHP_;
      v8f acc = {};
      acc = wmma16(qa0, load_tile16(Kb, DHP_, 0), acc);
      acc = wmma16(qa1, load_tile16(Kb, DHP_, 32), acc);
#pragma unroll
      for (int r = 0; r < 8; ++r)
        sS[(size_t)(r + 8 * hf) * T_ + nt * 16 + nn] = acc[r];
    }
    __syncthreads();

    // softmax over 512 per row (f32)
    float* srow = sS + (size_t)row * T_;
    float mx = -3.0e38f;
    for (int c = seg * 64; c < seg * 64 + 64; ++c) mx = fmaxf(mx, srow[c]);
    sRed[tid] = mx;
    __syncthreads();
    float rmax = sRed[row * 8];
#pragma unroll
    for (int j = 1; j < 8; ++j) rmax = fmaxf(rmax, sRed[row * 8 + j]);
    float s = 0.0f;
    for (int c = seg * 64; c < seg * 64 + 64; ++c) {
      float e = __expf(srow[c] - rmax);
      srow[c] = e;
      s += e;
    }
    __syncthreads();
    sRed[tid] = s;
    __syncthreads();
    float rsum = 0.0f;
#pragma unroll
    for (int j = 0; j < 8; ++j) rsum += sRed[row * 8 + j];
    float inv = 1.0f / rsum;
    for (int c = seg * 64; c < seg * 64 + 64; ++c) {
      float p = srow[c] * inv;
      sP[(size_t)row * T_ + c] = (_Float16)p;
      sMean[(size_t)row * T_ + c] += p * (1.0f / H_);
    }
    __syncthreads();

    // ctx(16x64) = P(16x512) @ V(512x64); wave w -> cols w*16..w*16+15
    {
      const half_t* Vb = VTl + (((size_t)(b * H_ + h)) * DHP_ + w * 16) * T_;
      v8f acc = {};
      for (int kc = 0; kc < T_ / 32; ++kc) {
        int k0 = kc * 32;
        v16h pa = load_tile16((const half_t*)sP, T_, k0);
        v16h vb = load_tile16(Vb, T_, k0);
        acc = wmma16(pa, vb, acc);
      }
#pragma unroll
      for (int r = 0; r < 8; ++r) {
        int m = r + 8 * hf;
        ctxp[((size_t)b * T_ + q0 + m) * CT_ + h * DHP_ + w * 16 + nn] = (half_t)acc[r];
      }
    }
    __syncthreads();
  }

  // masked = adj>0 ? leaky_relu(mean) : -9e15  (f32, f16 would overflow)
  for (int i = tid; i < 16 * T_; i += 128) {
    int mr = i >> 9, kc = i & (T_ - 1);
    float mv = sMean[i];
    float e = (mv > 0.0f) ? mv : 0.2f * mv;
    int a = adj[((size_t)b * T_ + q0 + mr) * T_ + kc];
    masked[((size_t)b * T_ + q0 + mr) * T_ + kc] = (a > 0) ? e : -9.0e15f;
  }
}

// ---- K4: out projection, stored transposed: outT[B,320,T] f32 ---------------
__global__ void out_proj(const half_t* __restrict__ ctxp, const half_t* __restrict__ WopT,
                         const float* __restrict__ bo, float* __restrict__ outT) {
  __shared__ __align__(16) half_t sWo[16 * CT_];  // 20 KB shared weight tile
  int b = blockIdx.z;
  int tid = threadIdx.x, w = tid >> 5;
  int t0 = (blockIdx.x * 4 + w) * 16;
  int n0 = blockIdx.y * 16;
  const half_t* Wg = WopT + (size_t)n0 * CT_;
  for (int i = tid * 8; i < 16 * CT_; i += 128 * 8)  // 10 x b128 per thread
    async_copy_b128(sWo + i, Wg + i);
  wait_asynccnt0();
  __syncthreads();

  const half_t* A = ctxp + ((size_t)b * T_ + t0) * CT_;
  v8f acc = {};
  for (int kc = 0; kc < CT_ / 32; ++kc) {
    int k0 = kc * 32;
    v16h a = load_tile16(A, CT_, k0);
    v16h bm = load_tile16(sWo, CT_, k0);
    acc = wmma16(a, bm, acc);
  }
  int lane = tid & 31, hf = lane >> 4, n = lane & 15;
  int col = n0 + n;
  float bs = (col < D_) ? bo[col] : 0.0f;
#pragma unroll
  for (int r = 0; r < 8; ++r) {
    int m = r + 8 * hf;
    outT[((size_t)b * NP_ + col) * T_ + (t0 + m)] = acc[r] + bs;
  }
}

// ---- K5: h_prime = masked(512x512 f32) @ out, fp32 WMMA (mask = -9e15) ------
__global__ void h_prime(const float* __restrict__ masked, const float* __restrict__ outT,
                        float* __restrict__ hp) {
  int b = blockIdx.z;
  int w = threadIdx.x >> 5;
  int nt = blockIdx.y * 2 + w;
  if (nt >= 19) return;  // cols 304..319 are pad; wave-uniform exit
  int q0 = blockIdx.x * 16;
  int n0 = nt * 16;
  const float* A  = masked + ((size_t)b * T_ + q0) * T_;
  const float* Bt = outT   + ((size_t)b * NP_ + n0) * T_;
  v8f acc = {};
  for (int k0 = 0; k0 < T_; k0 += 4) {
    v2f a  = load_tile4(A,  T_, k0);
    v2f bm = load_tile4(Bt, T_, k0);
    acc = __builtin_amdgcn_wmma_f32_16x16x4_f32(false, a, false, bm, (short)0, acc, false, false);
  }
  int lane = threadIdx.x & 31, hf = lane >> 4, n = lane & 15;
  int col = n0 + n;
#pragma unroll
  for (int r = 0; r < 8; ++r) {
    int m = r + 8 * hf;
    if (col < D_)
      hp[((size_t)b * T_ + q0 + m) * D_ + col] = acc[r];
  }
}

// ---- K6: y = LayerNorm(x + h_prime) -----------------------------------------
__global__ void layer_norm(const float* __restrict__ x, const float* __restrict__ hp,
                           const float* __restrict__ gamma, const float* __restrict__ beta,
                           float* __restrict__ out) {
  int row = blockIdx.x * 4 + (threadIdx.x >> 5);
  int lane = threadIdx.x & 31;
  const float* xr = x + (size_t)row * D_;
  const float* hr = hp + (size_t)row * D_;
  float s = 0.0f, sq = 0.0f;
  for (int c = lane; c < D_; c += 32) {
    float z = xr[c] + hr[c];
    s += z; sq += z * z;
  }
#pragma unroll
  for (int off = 16; off > 0; off >>= 1) {
    s  += __shfl_down(s,  off, 32);
    sq += __shfl_down(sq, off, 32);
  }
  s  = __shfl(s,  0, 32);
  sq = __shfl(sq, 0, 32);
  float mu  = s / (float)D_;
  float var = sq / (float)D_ - mu * mu;
  float inv = rsqrtf(var + 1e-5f);
  float* orow = out + (size_t)row * D_;
  for (int c = lane; c < D_; c += 32)
    orow[c] = (xr[c] + hr[c] - mu) * inv * gamma[c] + beta[c];
}

// ---- launch -----------------------------------------------------------------
extern "C" void kernel_launch(void* const* d_in, const int* in_sizes, int n_in,
                              void* d_out, int out_size, void* d_ws, size_t ws_size,
                              hipStream_t stream) {
  const float* x    = (const float*)d_in[0];
  const int*   adj  = (const int*)d_in[1];
  const int*   aidx = (const int*)d_in[2];
  const float* Wq   = (const float*)d_in[3];
  const float* bq   = (const float*)d_in[4];
  const float* Wk   = (const float*)d_in[5];
  const float* bk   = (const float*)d_in[6];
  const float* Wv   = (const float*)d_in[7];
  const float* bv   = (const float*)d_in[8];
  const float* Wo   = (const float*)d_in[9];
  const float* bo   = (const float*)d_in[10];
  const float* gam  = (const float*)d_in[11];
  const float* bet  = (const float*)d_in[12];
  float* out = (float*)d_out;

  char* ws = (char*)d_ws;
  size_t off = 0;
  half_t* tiH   = (half_t*)(ws + off); off += (size_t)B_ * T_ * KP_ * 2;            // 39.8 MB
  half_t* WTs   = (half_t*)(ws + off); off += (size_t)3 * H_ * DHP_ * KP_ * 2;      //  2.3 MB
  half_t* WopT  = (half_t*)(ws + off); off += (size_t)NP_ * CT_ * 2;                //  0.4 MB
  half_t* Qp    = (half_t*)(ws + off); off += (size_t)B_ * H_ * T_ * DHP_ * 2;      // 41.9 MB
  half_t* Kp    = (half_t*)(ws + off); off += (size_t)B_ * H_ * T_ * DHP_ * 2;      // 41.9 MB
  half_t* VTl   = (half_t*)(ws + off); off += (size_t)B_ * H_ * T_ * DHP_ * 2;      // 41.9 MB
  half_t* ctxp  = (half_t*)(ws + off); off += (size_t)B_ * T_ * CT_ * 2;            // 41.9 MB
  float*  maskd = (float*)(ws + off);  off += (size_t)B_ * T_ * T_ * 4;             // 67.1 MB
  // aliases: dead after the attention kernel completes (stream-ordered)
  float* hp   = (float*)Qp;   // needs 39.3 MB <= 41.9 MB
  float* outT = (float*)Kp;   // needs 41.9 MB == 41.9 MB

  {
    int tot = 3 * H_ * DHP_ * KP_ + NP_ * CT_;
    prep_weights<<<(tot + 255) / 256, 256, 0, stream>>>(Wq, Wk, Wv, Wo, WTs, WopT);
  }
  build_ti<<<dim3(T_, B_), 256, 0, stream>>>(x, aidx, tiH);
  qkv_gemm<<<dim3(T_ / 64, 3 * H_, B_), 128, 0, stream>>>(tiH, WTs, bq, bk, bv, Qp, Kp, VTl);
  attention<<<dim3(T_ / 16, B_), 128, 0, stream>>>(Qp, Kp, VTl, adj, ctxp, maskd);
  out_proj<<<dim3(T_ / 64, NP_ / 16, B_), 128, 0, stream>>>(ctxp, WopT, bo, outT);
  h_prime<<<dim3(T_ / 16, 10, B_), 64, 0, stream>>>(maskd, outT, hp);
  layer_norm<<<(B_ * T_) / 4, 128, 0, stream>>>(x, hp, gam, bet, out);
}
